// MambaBlock_82300163326304
// MI455X (gfx1250) — matrix operand order repaired and checked by reading, not compiled, not run on hardware
//
#include <hip/hip_runtime.h>
#include <hip/hip_bf16.h>
#include <stdint.h>

// ---------------- problem constants (fixed by reference) ----------------
#define BATCH   4
#define SEQL    2048
#define EMBD    1024
#define DINNER  2048          // N_EMBD * EXPAND
#define DPROJ   4096          // 2 * DINNER
#define NSTATE  16
#define DCONV   4
#define NROWS   (BATCH * SEQL)   // 8192 token rows
#define NXP     64               // padded x_proj output cols (33 -> 64)

typedef __attribute__((ext_vector_type(16))) _Float16 v16h;
typedef __attribute__((ext_vector_type(4)))  _Float16 v4h;
typedef __attribute__((ext_vector_type(8)))  float    v8f;

// =====================================================================
// WMMA GEMM: C(MxN) = A(MxK) * B(KxN); f32 in memory, f16 multiply /
// f32 accumulate. Block = 256 thr = 8 wave32. Block tile 128x64, BK=64.
// Wave tile 32x32: 4 accumulators, 2 A-frags x 2 B-frags per k-chunk ->
// 8 wmma per slab per wave from 8 fragment loads (2x reuse each side).
// LDS tiles stored PRE-PERMUTED into the wave32 WMMA register layout
// [kchunk][subtile][lane][e]: fragment = contiguous 32B (2x ds_load_b128).
// Register double-buffer of next slab + LDS ping-pong, 1 barrier/iter.
// =====================================================================
#define BM 128
#define BN 64
#define BK 64
#define BUFHA (2*8*32*16)   // halves per A LDS buffer (16 KB)
#define BUFHB (2*4*32*16)   // halves per B LDS buffer ( 8 KB)

__global__ __launch_bounds__(256)
void wmma_gemm_f16(const float* __restrict__ A, const float* __restrict__ B,
                   float* __restrict__ C, int M, int N, int K)
{
    __shared__ __align__(32) _Float16 Ap[2][2][8][32][16];  // [buf][kch][msub][lane][e]
    __shared__ __align__(32) _Float16 Bp[2][2][4][32][16];  // [buf][kch][nsub][lane][e]

    const int tid  = threadIdx.x;
    const int lane = tid & 31;
    const int wid  = tid >> 5;
    const int wm   = wid >> 1;          // 0..3 : pair of M sub-tiles (32 rows)
    const int wn   = wid & 1;           // 0..1 : pair of N sub-tiles (32 cols)
    const int m0   = blockIdx.y * BM;
    const int n0   = blockIdx.x * BN;

    _Float16* Abase = &Ap[0][0][0][0][0];
    _Float16* Bbase = &Bp[0][0][0][0][0];

    // ---- per-thread staging descriptors: A = 8 float4, B = 4 float4 ----
    const float* aP[8]; int aOff[8];
    const float* bP[4]; int bOff[4];
    #pragma unroll
    for (int j = 0; j < 8; ++j) {
        int i  = tid + j * 256;             // 0..2047
        int rr = i >> 4;                    // 0..127
        int cc = (i & 15) << 2;             // 0..60 step 4
        aP[j] = A + (size_t)(m0 + rr) * K + cc;
        int kch = cc >> 5, kl = cc & 31;
        int ha  = (kl >> 3) & 1;
        int eb  = (kl & 7) + ((kl >> 4) << 3);              // 0,4,8,12
        aOff[j] = (((kch * 8 + (rr >> 4)) * 32) + ((rr & 15) + 16 * ha)) * 16 + eb;
    }
    #pragma unroll
    for (int j = 0; j < 4; ++j) {
        int i  = tid + j * 256;             // 0..1023
        int kk = i >> 4;                    // 0..63
        int nn = (i & 15) << 2;             // 0..60 step 4
        bP[j] = B + (size_t)kk * N + n0 + nn;
        int kch = kk >> 5, kl = kk & 31;
        int hb = (kl >> 3) & 1, e = (kl & 7) + ((kl >> 4) << 3);
        bOff[j] = (((kch * 4 + (nn >> 4)) * 32) + ((nn & 15) + 16 * hb)) * 16 + e;
    }

    float4 av[8], bv[4];
    // ---- prologue: load + stage slab 0 ----
    #pragma unroll
    for (int j = 0; j < 8; ++j) av[j] = *(const float4*)(aP[j]);
    #pragma unroll
    for (int j = 0; j < 4; ++j) bv[j] = *(const float4*)(bP[j]);
    #pragma unroll
    for (int j = 0; j < 8; ++j) {
        v4h p; p[0] = (_Float16)av[j].x; p[1] = (_Float16)av[j].y;
               p[2] = (_Float16)av[j].z; p[3] = (_Float16)av[j].w;
        *(v4h*)(Abase + aOff[j]) = p;
    }
    #pragma unroll
    for (int j = 0; j < 4; ++j) {
        _Float16* d = Bbase + bOff[j];
        d[0]  = (_Float16)bv[j].x; d[16] = (_Float16)bv[j].y;
        d[32] = (_Float16)bv[j].z; d[48] = (_Float16)bv[j].w;
    }
    __syncthreads();

    v8f acc00 = {}, acc01 = {}, acc10 = {}, acc11 = {};
    int buf = 0;

    for (int kt = 0; kt < K; kt += BK) {
        const bool more = (kt + BK) < K;
        if (more) {   // register prefetch of next slab (overlaps with wmma)
            #pragma unroll
            for (int j = 0; j < 8; ++j)
                av[j] = *(const float4*)(aP[j] + (kt + BK));
            #pragma unroll
            for (int j = 0; j < 4; ++j)
                bv[j] = *(const float4*)(bP[j] + (size_t)(kt + BK) * N);
        }
        // ---- compute from current buffer: 2 k-chunks x (2x2) fragment outer product ----
        const _Float16* Ab = Abase + buf * BUFHA;
        const _Float16* Bb = Bbase + buf * BUFHB;
        #pragma unroll
        for (int kc = 0; kc < 2; ++kc) {
            v16h a0 = *(const v16h*)(Ab + ((kc * 8 + wm * 2 + 0) * 32 + lane) * 16);
            v16h a1 = *(const v16h*)(Ab + ((kc * 8 + wm * 2 + 1) * 32 + lane) * 16);
            v16h b0 = *(const v16h*)(Bb + ((kc * 4 + wn * 2 + 0) * 32 + lane) * 16);
            v16h b1 = *(const v16h*)(Bb + ((kc * 4 + wn * 2 + 1) * 32 + lane) * 16);
            acc00 = __builtin_amdgcn_wmma_f32_16x16x32_f16(false, a0, false, b0,
                                                           (short)0, acc00, false, false);
            acc01 = __builtin_amdgcn_wmma_f32_16x16x32_f16(false, a0, false, b1,
                                                           (short)0, acc01, false, false);
            acc10 = __builtin_amdgcn_wmma_f32_16x16x32_f16(false, a1, false, b0,
                                                           (short)0, acc10, false, false);
            acc11 = __builtin_amdgcn_wmma_f32_16x16x32_f16(false, a1, false, b1,
                                                           (short)0, acc11, false, false);
        }
        // ---- stage next slab into the other buffer ----
        if (more) {
            _Float16* An = Abase + (buf ^ 1) * BUFHA;
            _Float16* Bn = Bbase + (buf ^ 1) * BUFHB;
            #pragma unroll
            for (int j = 0; j < 8; ++j) {
                v4h p; p[0] = (_Float16)av[j].x; p[1] = (_Float16)av[j].y;
                       p[2] = (_Float16)av[j].z; p[3] = (_Float16)av[j].w;
                *(v4h*)(An + aOff[j]) = p;
            }
            #pragma unroll
            for (int j = 0; j < 4; ++j) {
                _Float16* d = Bn + bOff[j];
                d[0]  = (_Float16)bv[j].x; d[16] = (_Float16)bv[j].y;
                d[32] = (_Float16)bv[j].z; d[48] = (_Float16)bv[j].w;
            }
        }
        __syncthreads();
        buf ^= 1;
    }

    // ---- epilogue: C layout m = g + 8*half, n = lane&15 ----
    const int r = lane & 15, hf = lane >> 4;
    #pragma unroll
    for (int g = 0; g < 8; ++g) {
        size_t mA = (size_t)(m0 + (wm * 2 + 0) * 16 + hf * 8 + g);
        size_t mB = (size_t)(m0 + (wm * 2 + 1) * 16 + hf * 8 + g);
        int nA = n0 + (wn * 2 + 0) * 16 + r;
        int nB = n0 + (wn * 2 + 1) * 16 + r;
        C[mA * N + nA] = acc00[g];
        C[mA * N + nB] = acc01[g];
        C[mB * N + nA] = acc10[g];
        C[mB * N + nB] = acc11[g];
    }
}

// ---------------- depthwise causal conv(4) + bias + SiLU, fused with
// SiLU(res) from the other half of xr ----------
__device__ __forceinline__ float silu_f(float v) {
    return v / (1.0f + __expf(-v));
}

__global__ __launch_bounds__(256)
void conv_silu_res(const float* __restrict__ xr,
                   const float* __restrict__ conv_w, const float* __restrict__ conv_b,
                   float* __restrict__ xcv, float* __restrict__ sres)
{
    int idx = blockIdx.x * blockDim.x + threadIdx.x;     // over NROWS*DINNER
    int d = idx & (DINNER - 1);
    int m = idx >> 11;                                   // token row (b*L + t)
    int t = m & (SEQL - 1);

    float acc = conv_b[d];
    #pragma unroll
    for (int j = 0; j < DCONV; ++j) {
        int tt = t - (DCONV - 1) + j;
        if (tt >= 0)
            acc += conv_w[d * DCONV + j] * xr[(size_t)(m + tt - t) * DPROJ + d];
    }
    xcv[(size_t)idx] = silu_f(acc);

    float rv = xr[(size_t)m * DPROJ + DINNER + d];
    sres[(size_t)idx] = silu_f(rv);
}

// ---------------- pad x_proj_w (2048x33) -> (2048x64) with zeros so the
// skinny projection also runs on the WMMA pipe ----------
__global__ __launch_bounds__(256)
void pad_xproj_w(const float* __restrict__ w, float* __restrict__ wpad)
{
    int idx = blockIdx.x * blockDim.x + threadIdx.x;     // DINNER * NXP
    int k = idx >> 6, c = idx & 63;
    wpad[idx] = (c < 33) ? w[k * 33 + c] : 0.f;
}

// ---------------- split x_dbl (NROWS x 64) into dt(1), B(16), C(16) ----------
__global__ __launch_bounds__(256)
void split_dbc(const float* __restrict__ dbl,
               float* __restrict__ dtv, float* __restrict__ Bv, float* __restrict__ Cv)
{
    int idx = blockIdx.x * blockDim.x + threadIdx.x;     // NROWS * 64
    int m = idx >> 6, c = idx & 63;
    if (c >= 33) return;
    float v = dbl[idx];
    if (c == 0)       dtv[m] = v;
    else if (c < 17)  Bv[m * NSTATE + (c - 1)]  = v;
    else              Cv[m * NSTATE + (c - 17)] = v;
}

// ---------------- selective scan: one thread per (batch, channel); 16-state
// recurrence in VGPRs; dt/B/C loads block-uniform (scalarizable); fuses
// y * silu(res) into the output so out_proj reads a single tensor. ----------
__global__ __launch_bounds__(256)
void scan_kernel(const float* __restrict__ xcv, const float* __restrict__ dtv,
                 const float* __restrict__ Bv,  const float* __restrict__ Cv,
                 const float* __restrict__ sres,
                 const float* __restrict__ dtw, const float* __restrict__ dtb,
                 const float* __restrict__ A_log, const float* __restrict__ Dp,
                 float* __restrict__ yf)
{
    int b = blockIdx.x >> 3;                               // 4 batches
    int d = ((blockIdx.x & 7) << 8) + threadIdx.x;         // 2048 channels

    float Arow[NSTATE], h[NSTATE];
    #pragma unroll
    for (int n = 0; n < NSTATE; ++n) {
        Arow[n] = -__expf(A_log[d * NSTATE + n]);
        h[n] = 0.f;
    }
    const float wdt = dtw[d], bdt = dtb[d], Dd = Dp[d];
    const size_t rbase = (size_t)b * SEQL;

    for (int t = 0; t < SEQL; ++t) {
        const size_t row = rbase + t;
        float xv   = xcv[row * DINNER + d];
        float praw = dtv[row] * wdt + bdt;
        float delta = (praw > 20.f) ? praw : __logf(1.f + __expf(praw));  // softplus
        float dx = delta * xv;
        float y  = Dd * xv;
        #pragma unroll
        for (int n = 0; n < NSTATE; ++n) {
            float abar = __expf(delta * Arow[n]);
            h[n] = abar * h[n] + Bv[row * NSTATE + n] * dx;
            y   += h[n] * Cv[row * NSTATE + n];
        }
        yf[row * DINNER + d] = y * sres[row * DINNER + d];
    }
}

// ---------------- launcher ----------------
extern "C" void kernel_launch(void* const* d_in, const int* in_sizes, int n_in,
                              void* d_out, int out_size, void* d_ws, size_t ws_size,
                              hipStream_t stream)
{
    const float* x         = (const float*)d_in[0];
    const float* in_proj_w = (const float*)d_in[1];
    const float* conv_w    = (const float*)d_in[2];
    const float* conv_b    = (const float*)d_in[3];
    const float* x_proj_w  = (const float*)d_in[4];
    const float* dt_proj_w = (const float*)d_in[5];
    const float* dt_proj_b = (const float*)d_in[6];
    const float* A_log     = (const float*)d_in[7];
    const float* Dp        = (const float*)d_in[8];
    const float* out_proj_w= (const float*)d_in[9];
    float* out = (float*)d_out;

    char* ws = (char*)d_ws;
    size_t off = 0;
    float* xr   = (float*)(ws + off); off += (size_t)NROWS * DPROJ  * 4;  // 128 MB
    float* xcv  = (float*)(ws + off); off += (size_t)NROWS * DINNER * 4;  //  64 MB
    float* sres = (float*)(ws + off); off += (size_t)NROWS * DINNER * 4;  //  64 MB
    float* yf   = (float*)(ws + off); off += (size_t)NROWS * DINNER * 4;  //  64 MB
    float* wpad = (float*)(ws + off); off += (size_t)DINNER * NXP  * 4;   // 512 KB
    float* dbl  = (float*)(ws + off); off += (size_t)NROWS * NXP   * 4;   //   2 MB
    float* dtv  = (float*)(ws + off); off += (size_t)NROWS * 4;
    float* Bv   = (float*)(ws + off); off += (size_t)NROWS * NSTATE * 4;
    float* Cv   = (float*)(ws + off); off += (size_t)NROWS * NSTATE * 4;

    // 1) in_proj GEMM: (8192x1024) @ (1024x4096) -> xr
    {
        dim3 grid(DPROJ / BN, NROWS / BM);
        wmma_gemm_f16<<<grid, 256, 0, stream>>>(x, in_proj_w, xr, NROWS, DPROJ, EMBD);
    }
    // 2) depthwise conv + SiLU, plus SiLU(res)
    conv_silu_res<<<(NROWS * DINNER) / 256, 256, 0, stream>>>(xr, conv_w, conv_b, xcv, sres);
    // 3) x_proj on the WMMA pipe: pad weights, GEMM, split
    pad_xproj_w<<<(DINNER * NXP) / 256, 256, 0, stream>>>(x_proj_w, wpad);
    {
        dim3 grid(NXP / BN, NROWS / BM);
        wmma_gemm_f16<<<grid, 256, 0, stream>>>(xcv, wpad, dbl, NROWS, NXP, DINNER);
    }
    split_dbc<<<(NROWS * NXP) / 256, 256, 0, stream>>>(dbl, dtv, Bv, Cv);
    // 4) selective scan (fuses y * silu(res))
    scan_kernel<<<BATCH * (DINNER / 256), 256, 0, stream>>>(
        xcv, dtv, Bv, Cv, sres, dt_proj_w, dt_proj_b, A_log, Dp, yf);
    // 5) out_proj GEMM: (8192x2048) @ (2048x1024) -> out
    {
        dim3 grid(EMBD / BN, NROWS / BM);
        wmma_gemm_f16<<<grid, 256, 0, stream>>>(yf, out_proj_w, out, NROWS, EMBD, DINNER);
    }
}
